// GATLayer_35313221108351
// MI455X (gfx1250) — matrix operand order repaired and checked by reading, not compiled, not run on hardware
//
#include <hip/hip_runtime.h>
#include <hip/hip_bf16.h>

typedef __attribute__((ext_vector_type(16))) _Float16 v16h;
typedef __attribute__((ext_vector_type(8)))  _Float16 v8h;
typedef __attribute__((ext_vector_type(8)))  float    v8f;

#define V_N     50000
#define E_N     800000
#define NODE_IN 128
#define EDGE_IN 16
#define OUTF    32
#define HEADS   4
#define HF      128            // HEADS*OUTF
#define NEG_SLOPE 0.2f

// ---- workspace layout (float offsets) ----
#define OFF_FT    0                                  // V*128
#define OFF_LG    (OFF_FT   + V_N*HF)                // E*4 (logits, then reused as exp)
#define OFF_EL    (OFF_LG   + E_N*HEADS)             // V*4
#define OFF_ER    (OFF_EL   + V_N*HEADS)             // V*4
#define OFF_VMAX  (OFF_ER   + V_N*HEADS)             // V*4
#define OFF_ESUM  (OFF_VMAX + V_N*HEADS)             // V*4
#define OFF_WAL   (OFF_ESUM + V_N*HEADS)             // 128*4
#define OFF_WAR   (OFF_WAL  + NODE_IN*HEADS)         // 128*4
#define OFF_WEA   (OFF_WAR  + NODE_IN*HEADS)         // 16*4
#define OFF_PACKB (OFF_WEA  + 64)                    // 64 frags * 512 halfs (32B-aligned)

__device__ __forceinline__ float atomicMaxF(float* addr, float val) {
    // works for mixed signs given -inf init (standard int/uint ordering trick)
    if (val >= 0.0f)
        return __int_as_float(atomicMax((int*)addr, __float_as_int(val)));
    else
        return __uint_as_float(atomicMin((unsigned int*)addr, __float_as_uint(val)));
}

// ---- fold attention vectors through the projection weights ----
// wal[k][h] = sum_f W_node[k][h*32+f]*attn_l[h][f]   (same for war / wea)
__global__ __launch_bounds__(512) void eg_fold(
    const float* __restrict__ Wn, const float* __restrict__ We,
    const float* __restrict__ al, const float* __restrict__ ar,
    const float* __restrict__ ae,
    float* __restrict__ wal, float* __restrict__ war, float* __restrict__ wea)
{
    const int t = threadIdx.x;          // 0..511
    const int k = t >> 2, h = t & 3;
    float s1 = 0.f, s2 = 0.f;
    #pragma unroll
    for (int f = 0; f < OUTF; ++f) {
        const float w = Wn[k*HF + h*OUTF + f];
        s1 += w * al[h*OUTF + f];
        s2 += w * ar[h*OUTF + f];
    }
    wal[t] = s1;  war[t] = s2;
    if (t < EDGE_IN*HEADS) {
        float s3 = 0.f;
        #pragma unroll
        for (int f = 0; f < OUTF; ++f)
            s3 += We[k*HF + h*OUTF + f] * ae[h*OUTF + f];
        wea[t] = s3;
    }
}

// ---- pack f16 B fragments for [128 x 256] concat(W_node, res_W) ----
// frag (kc,nt): K = kc*32 + (i&7) + 16*(i>>3) + (lane>=16 ? 8 : 0) ; n = nt*16 + (lane&15)
__global__ __launch_bounds__(512) void eg_pack(
    const float* __restrict__ Wn, const float* __restrict__ Wr,
    _Float16* __restrict__ packB)
{
    const int gid  = blockIdx.x * 512 + threadIdx.x;   // 64*512 total
    const int fidx = gid >> 9;
    const int idx  = gid & 511;
    const int lane = idx >> 4, i = idx & 15;
    const int kc = fidx >> 4, nt = fidx & 15;
    const int k = kc*32 + (i & 7) + ((i >> 3) << 4) + ((lane < 16) ? 0 : 8);
    const int n = nt*16 + (lane & 15);
    const float w = (n < HF) ? Wn[k*HF + n] : Wr[k*HF + (n - HF)];
    packB[gid] = (_Float16)w;
}

__global__ void eg_init(float* __restrict__ vmax, float* __restrict__ esum) {
    const int i = blockIdx.x * blockDim.x + threadIdx.x;
    if (i < V_N*HEADS) { vmax[i] = -INFINITY; esum[i] = 0.f; }
}

// ---- fused node GEMM: ft = node@W_node -> ws ; out = node@res_W + bias ----
__global__ __launch_bounds__(128) void eg_node_gemm(
    const float* __restrict__ nf, const _Float16* __restrict__ packB,
    const float* __restrict__ bias,
    float* __restrict__ ft, float* __restrict__ out)
{
    __shared__ alignas(16) _Float16 sA[16*NODE_IN];   // 4KB tile, f16
    const int t    = threadIdx.x;
    const int tile = blockIdx.x;                      // V_N/16 = 3125 exact
    #pragma unroll
    for (int j = 0; j < 16; ++j)
        sA[j*NODE_IN + t] = (_Float16)nf[(size_t)(tile*16 + j)*NODE_IN + t];
    __syncthreads();

    const int wave = t >> 5, lane = t & 31;
    const int r  = lane & 15;
    const int hi = (lane >= 16);

    // build A fragments for the 4 K-chunks (documented 16-bit A layout)
    v16h a[4];
    #pragma unroll
    for (int kc = 0; kc < 4; ++kc) {
        const int base = r*NODE_IN + kc*32 + (hi ? 8 : 0);
        v8h s0 = *(const v8h*)&sA[base];       // K {0..7}  (or {8..15})
        v8h s1 = *(const v8h*)&sA[base + 16];  // K {16..23}(or {24..31})
        #pragma unroll
        for (int i = 0; i < 8; ++i) { a[kc][i] = s0[i]; a[kc][i+8] = s1[i]; }
    }

    const int n_    = lane & 15;
    const int mbase = tile*16 + (hi ? 8 : 0);
    #pragma unroll
    for (int g = 0; g < 4; ++g) {
        const int nt = wave + g*4;             // 0..15 across 4 waves
        v8f acc = {};
        #pragma unroll
        for (int kc = 0; kc < 4; ++kc) {
            v16h b = *(const v16h*)(packB + ((size_t)(kc*16 + nt)*512 + lane*16));
            acc = __builtin_amdgcn_wmma_f32_16x16x32_f16(
                false, a[kc], false, b, (short)0, acc, false, false);
        }
        if (nt < 8) {                          // ft tile
            #pragma unroll
            for (int v = 0; v < 8; ++v)
                ft[(size_t)(mbase + v)*HF + nt*16 + n_] = acc[v];
        } else {                               // residual tile -> initialize out
            const int col = (nt - 8)*16 + n_;
            const float bb = bias[col];
            #pragma unroll
            for (int v = 0; v < 8; ++v)
                out[(size_t)(mbase + v)*HF + col] = acc[v] + bb;
        }
    }
}

// ---- el/er via folded [128x4] weights; one node per wave ----
__global__ __launch_bounds__(256) void eg_node_attn(
    const float* __restrict__ nf,
    const float* __restrict__ wal, const float* __restrict__ war,
    float* __restrict__ el, float* __restrict__ er)
{
    const int lane = threadIdx.x & 31;
    const int wave = threadIdx.x >> 5;
    const int v = blockIdx.x*8 + wave;         // V_N/8 = 6250 exact
    float al4[4] = {0,0,0,0}, ar4[4] = {0,0,0,0};
    #pragma unroll
    for (int j = 0; j < 4; ++j) {
        const int k = lane + 32*j;
        const float x = nf[(size_t)v*NODE_IN + k];
        #pragma unroll
        for (int h = 0; h < 4; ++h) {
            al4[h] += x * wal[k*4 + h];
            ar4[h] += x * war[k*4 + h];
        }
    }
    #pragma unroll
    for (int off = 16; off > 0; off >>= 1) {
        #pragma unroll
        for (int h = 0; h < 4; ++h) {
            al4[h] += __shfl_xor(al4[h], off, 32);
            ar4[h] += __shfl_xor(ar4[h], off, 32);
        }
    }
    if (lane == 0) {
        #pragma unroll
        for (int h = 0; h < 4; ++h) { el[v*4 + h] = al4[h]; er[v*4 + h] = ar4[h]; }
    }
}

// ---- per-edge logits + segment max ----
__global__ void eg_edge_logits(
    const float* __restrict__ ef,
    const int* __restrict__ src, const int* __restrict__ dst,
    const float* __restrict__ wea,
    const float* __restrict__ el, const float* __restrict__ er,
    float* __restrict__ lg, float* __restrict__ vmax)
{
    const int e = blockIdx.x * blockDim.x + threadIdx.x;
    if (e >= E_N) return;
    const float4* ep = (const float4*)(ef + (size_t)e*EDGE_IN);
    float4 q0 = ep[0], q1 = ep[1], q2 = ep[2], q3 = ep[3];
    const float x[16] = {q0.x,q0.y,q0.z,q0.w, q1.x,q1.y,q1.z,q1.w,
                         q2.x,q2.y,q2.z,q2.w, q3.x,q3.y,q3.z,q3.w};
    const int s = src[e], d = dst[e];
    const float4 e4 = ((const float4*)el)[s];
    const float4 r4 = ((const float4*)er)[d];
    const float elv[4] = {e4.x,e4.y,e4.z,e4.w};
    const float erv[4] = {r4.x,r4.y,r4.z,r4.w};
    #pragma unroll
    for (int h = 0; h < 4; ++h) {
        float acc = elv[h] + erv[h];
        #pragma unroll
        for (int k = 0; k < 16; ++k) acc += x[k] * wea[k*4 + h];
        const float l = (acc >= 0.f) ? acc : NEG_SLOPE*acc;
        lg[(size_t)e*4 + h] = l;
        atomicMaxF(&vmax[(size_t)d*4 + h], l);
    }
}

// ---- exp(logit - max) + segment sum ----
__global__ void eg_edge_exp(
    const int* __restrict__ dst,
    float* __restrict__ lg, const float* __restrict__ vmax,
    float* __restrict__ esum)
{
    const int e = blockIdx.x * blockDim.x + threadIdx.x;
    if (e >= E_N) return;
    const int d = dst[e];
    const float4 m4 = ((const float4*)vmax)[d];
    const float m[4] = {m4.x, m4.y, m4.z, m4.w};
    #pragma unroll
    for (int h = 0; h < 4; ++h) {
        const float ex = __expf(lg[(size_t)e*4 + h] - m[h]);
        lg[(size_t)e*4 + h] = ex;
        atomicAdd(&esum[(size_t)d*4 + h], ex);
    }
}

// ---- weighted message scatter: out[dst] += ft[src] * a ----
__global__ __launch_bounds__(128) void eg_scatter(
    const float* __restrict__ ft,
    const int* __restrict__ src, const int* __restrict__ dst,
    const float* __restrict__ lg, const float* __restrict__ esum,
    float* __restrict__ out)
{
    const int e = blockIdx.x;        // one edge per block, 128 cols
    const int t = threadIdx.x;
    const int s = src[e], d = dst[e];
    const int h = t >> 5;            // head = col/32
    const float a = lg[(size_t)e*4 + h] / esum[(size_t)d*4 + h];
    atomicAdd(&out[(size_t)d*HF + t], ft[(size_t)s*HF + t] * a);
}

extern "C" void kernel_launch(void* const* d_in, const int* in_sizes, int n_in,
                              void* d_out, int out_size, void* d_ws, size_t ws_size,
                              hipStream_t stream) {
    const float* nf   = (const float*)d_in[0];
    const float* ef   = (const float*)d_in[1];
    const int*   src  = (const int*)  d_in[2];
    const int*   dst  = (const int*)  d_in[3];
    const float* Wn   = (const float*)d_in[4];
    const float* We   = (const float*)d_in[5];
    const float* al   = (const float*)d_in[6];
    const float* ar   = (const float*)d_in[7];
    const float* ae   = (const float*)d_in[8];
    const float* Wr   = (const float*)d_in[9];
    const float* bias = (const float*)d_in[10];
    float* out = (float*)d_out;

    float* wsf   = (float*)d_ws;
    float* ft    = wsf + OFF_FT;
    float* lg    = wsf + OFF_LG;
    float* el    = wsf + OFF_EL;
    float* er    = wsf + OFF_ER;
    float* vmax  = wsf + OFF_VMAX;
    float* esum  = wsf + OFF_ESUM;
    float* wal   = wsf + OFF_WAL;
    float* war   = wsf + OFF_WAR;
    float* wea   = wsf + OFF_WEA;
    _Float16* packB = (_Float16*)(wsf + OFF_PACKB);

    eg_fold<<<1, 512, 0, stream>>>(Wn, We, al, ar, ae, wal, war, wea);
    eg_pack<<<64, 512, 0, stream>>>(Wn, Wr, packB);
    eg_init<<<(V_N*HEADS + 255)/256, 256, 0, stream>>>(vmax, esum);
    eg_node_gemm<<<V_N/16, 128, 0, stream>>>(nf, packB, bias, ft, out);
    eg_node_attn<<<V_N/8, 256, 0, stream>>>(nf, wal, war, el, er);
    eg_edge_logits<<<(E_N + 255)/256, 256, 0, stream>>>(ef, src, dst, wea, el, er, lg, vmax);
    eg_edge_exp<<<(E_N + 255)/256, 256, 0, stream>>>(dst, lg, vmax, esum);
    eg_scatter<<<E_N, 128, 0, stream>>>(ft, src, dst, lg, esum, out);
}
